// MHSABlock_24111946399781
// MI455X (gfx1250) — compile-verified, matrix operand order checked by reading, not run on hardware
//
#include <hip/hip_runtime.h>

typedef __attribute__((ext_vector_type(16))) __bf16       v16bf;
typedef __attribute__((ext_vector_type(8)))  float        v8f;
typedef __attribute__((ext_vector_type(4)))  unsigned int u4;

// Load one 16x16x32 bf16 WMMA operand fragment (A-layout / B-as-[n][k]-layout)
// from an LDS row starting at p: elements [0..7] and [16..23] relative to the
// lane's K-half offset (caller bakes kh into p).
__device__ __forceinline__ v16bf ldfrag(const __bf16* p) {
  union { v16bf v; u4 u[2]; } f;
  f.u[0] = *(const u4*)(p);
  f.u[1] = *(const u4*)(p + 16);
  return f.v;
}

// ---------------------------------------------------------------------------
// Weight transpose + f32->bf16 convert:  in[R][C] f32  ->  out[C][R] bf16
// ---------------------------------------------------------------------------
__global__ __launch_bounds__(256) void trcvt_k(const float* __restrict__ in,
                                               __bf16* __restrict__ out,
                                               int R, int C) {
  __shared__ float t[32][33];
  const int tx = threadIdx.x, ty = threadIdx.y;
  const int r0 = blockIdx.y * 32, c0 = blockIdx.x * 32;
#pragma unroll
  for (int q = 0; q < 4; ++q)
    t[ty + q * 8][tx] = in[(size_t)(r0 + ty + q * 8) * C + c0 + tx];
  __syncthreads();
#pragma unroll
  for (int q = 0; q < 4; ++q)
    out[(size_t)(c0 + ty + q * 8) * R + r0 + tx] = (__bf16)t[tx][ty + q * 8];
}

// ---------------------------------------------------------------------------
// LayerNorm over C=1024, writes bf16. One block (256 thr) per row.
// ---------------------------------------------------------------------------
__global__ __launch_bounds__(256) void ln_bf16_k(const float* __restrict__ x,
                                                 const float* __restrict__ g,
                                                 const float* __restrict__ b,
                                                 __bf16* __restrict__ out) {
  __shared__ float red[256];
  const int row = blockIdx.x, tid = threadIdx.x;
  const float* xr = x + (size_t)row * 1024;
  float v[4];
  float s = 0.f;
#pragma unroll
  for (int i = 0; i < 4; ++i) { v[i] = xr[tid + i * 256]; s += v[i]; }
  red[tid] = s; __syncthreads();
  for (int off = 128; off > 0; off >>= 1) {
    if (tid < off) red[tid] += red[tid + off];
    __syncthreads();
  }
  const float mu = red[0] * (1.f / 1024.f);
  __syncthreads();
  s = 0.f;
#pragma unroll
  for (int i = 0; i < 4; ++i) { v[i] -= mu; s += v[i] * v[i]; }
  red[tid] = s; __syncthreads();
  for (int off = 128; off > 0; off >>= 1) {
    if (tid < off) red[tid] += red[tid + off];
    __syncthreads();
  }
  const float rstd = rsqrtf(red[0] * (1.f / 1024.f) + 1e-5f);
  __bf16* orow = out + (size_t)row * 1024;
#pragma unroll
  for (int i = 0; i < 4; ++i) {
    const int c = tid + i * 256;
    orow[c] = (__bf16)(v[i] * rstd * g[c] + b[c]);
  }
}

// ---------------------------------------------------------------------------
// bf16 WMMA GEMM: C[M,N] = A[M,K] * BT[N,K]^T (+ epilogue).
// Block tile 128x128, K-step 64, 8 waves (2x4), wave tile 64x32.
// Double-buffered LDS with register prefetch: ONE barrier per K-step; global
// loads for tile i+1 are issued before the WMMAs of tile i (latency hidden).
// L2 warmed two tiles ahead via global_prefetch.
// MODE 0: store bf16.  MODE 1: f32 = res + C + bias.  MODE 2: bf16 = GELU(C+bias).
// ---------------------------------------------------------------------------
template <int MODE>
__global__ __launch_bounds__(256) void gemm_bf16_k(
    const __bf16* __restrict__ A, const __bf16* __restrict__ BT,
    const float* __restrict__ bias, const float* __restrict__ res,
    void* __restrict__ outp, int M, int N, int K) {
  constexpr int LD = 72;  // 64 data + 8 pad (keeps 16B align, breaks conflicts)
  __shared__ __bf16 As[2][128 * LD];
  __shared__ __bf16 Bs[2][128 * LD];
  const int tid = threadIdx.x, lane = tid & 31, wid = tid >> 5;
  const int rw = wid >> 2, cw = wid & 3;
  const int m0 = blockIdx.y * 128, n0 = blockIdx.x * 128;

  // Per-thread staging coordinates (constant across K-steps).
  int row4[4], seg4[4];
#pragma unroll
  for (int rep = 0; rep < 4; ++rep) {
    const int idx = tid + rep * 256;  // 1024 u4 per matrix tile
    row4[rep] = idx >> 3;
    seg4[rep] = (idx & 7) << 3;
  }

  v8f acc[4][2] = {};
  u4 ra[4], rb[4];
  // Prologue: fetch tile ks=0 into registers.
#pragma unroll
  for (int rep = 0; rep < 4; ++rep) {
    ra[rep] = *(const u4*)&A[(size_t)(m0 + row4[rep]) * K + seg4[rep]];
    rb[rep] = *(const u4*)&BT[(size_t)(n0 + row4[rep]) * K + seg4[rep]];
  }

  int buf = 0;
  for (int ks = 0; ks < K; ks += 64) {
    // Commit prefetched tile to LDS buffer `buf`.
#pragma unroll
    for (int rep = 0; rep < 4; ++rep) {
      *(u4*)&As[buf][row4[rep] * LD + seg4[rep]] = ra[rep];
      *(u4*)&Bs[buf][row4[rep] * LD + seg4[rep]] = rb[rep];
    }
    __syncthreads();

    // Issue global loads for the next tile (overlap with WMMAs below).
    if (ks + 64 < K) {
#pragma unroll
      for (int rep = 0; rep < 4; ++rep) {
        ra[rep] = *(const u4*)&A[(size_t)(m0 + row4[rep]) * K + ks + 64 + seg4[rep]];
        rb[rep] = *(const u4*)&BT[(size_t)(n0 + row4[rep]) * K + ks + 64 + seg4[rep]];
      }
    }
    // Warm L2 two tiles ahead (global_prefetch_b8).
    if (ks + 128 < K) {
#pragma unroll
      for (int rep = 0; rep < 4; ++rep) {
        __builtin_prefetch(&A[(size_t)(m0 + row4[rep]) * K + ks + 128 + seg4[rep]], 0, 1);
        __builtin_prefetch(&BT[(size_t)(n0 + row4[rep]) * K + ks + 128 + seg4[rep]], 0, 1);
      }
    }

    const __bf16* Ap = As[buf];
    const __bf16* Bp = Bs[buf];
#pragma unroll
    for (int kk = 0; kk < 2; ++kk) {
      const int kh = kk * 32 + ((lane >> 4) << 3);
      v16bf av[4], bv[2];
#pragma unroll
      for (int i = 0; i < 4; ++i)
        av[i] = ldfrag(&Ap[(rw * 64 + i * 16 + (lane & 15)) * LD + kh]);
#pragma unroll
      for (int j = 0; j < 2; ++j)
        bv[j] = ldfrag(&Bp[(cw * 32 + j * 16 + (lane & 15)) * LD + kh]);
#pragma unroll
      for (int i = 0; i < 4; ++i)
#pragma unroll
        for (int j = 0; j < 2; ++j)
          acc[i][j] = __builtin_amdgcn_wmma_f32_16x16x32_bf16(
              false, av[i], false, bv[j], (short)0, acc[i][j], false, false);
    }
    buf ^= 1;
  }

#pragma unroll
  for (int i = 0; i < 4; ++i)
#pragma unroll
    for (int j = 0; j < 2; ++j) {
      const int col = n0 + cw * 32 + j * 16 + (lane & 15);
      const int rb2 = m0 + rw * 64 + i * 16 + ((lane >> 4) << 3);
#pragma unroll
      for (int rr = 0; rr < 8; ++rr) {
        const size_t o = (size_t)(rb2 + rr) * N + col;
        const float v = acc[i][j][rr];
        if constexpr (MODE == 0) {
          ((__bf16*)outp)[o] = (__bf16)v;
        } else if constexpr (MODE == 1) {
          ((float*)outp)[o] = res[o] + v + bias[col];
        } else {
          const float t = v + bias[col];
          ((__bf16*)outp)[o] = (__bf16)(0.5f * t * (1.0f + erff(t * 0.70710678118f)));
        }
      }
    }
}

// ---------------------------------------------------------------------------
// Flash attention, one block per (q-tile of 128 rows, b*16+h).
// qkv bf16 [B*2048][3072]; out bf16 [B*2048][1024].
// ---------------------------------------------------------------------------
__global__ __launch_bounds__(256) void attn_k(const __bf16* __restrict__ qkv,
                                              __bf16* __restrict__ out) {
  constexpr int LDQ = 72, LDS_S = 132, LDP = 136, LDV = 136;
  __shared__ __bf16 Qs[128 * LDQ];
  __shared__ __bf16 Ks[128 * LDQ];
  __shared__ __bf16 Vt[64 * LDV];
  __shared__ float  Sb[128 * LDS_S];
  __shared__ __bf16 Pb[128 * LDP];
  __shared__ float mrow[128], lrow[128], arow[128];

  const int tid = threadIdx.x, lane = tid & 31, wid = tid >> 5;
  const int qt = blockIdx.x, bh = blockIdx.y;
  const int b = bh >> 4, h = bh & 15;
  const size_t tokbase = (size_t)b * 2048;

  // Stage Q tile (128 x 64)
#pragma unroll
  for (int rep = 0; rep < 4; ++rep) {
    const int idx = tid + rep * 256;
    const int row = idx >> 3, seg = (idx & 7) << 3;
    *(u4*)&Qs[row * LDQ + seg] =
        *(const u4*)&qkv[(tokbase + qt * 128 + row) * 3072 + h * 64 + seg];
  }
  if (tid < 128) { mrow[tid] = -1e30f; lrow[tid] = 0.f; }

  const int rS = wid >> 2, cS = wid & 3;  // S: 2x4 waves, each 64x32
  const int rO = wid >> 1, cO = wid & 1;  // O: 4x2 waves, each 32x32
  v8f oacc[2][2] = {};

  for (int kb = 0; kb < 16; ++kb) {
    // Stage K block (128x64) and V^T block (64x128)
#pragma unroll
    for (int rep = 0; rep < 4; ++rep) {
      const int idx = tid + rep * 256;
      const int row = idx >> 3, seg = (idx & 7) << 3;
      const size_t base = (tokbase + kb * 128 + row) * 3072 + h * 64;
      *(u4*)&Ks[row * LDQ + seg] = *(const u4*)&qkv[base + 1024 + seg];
      union { u4 q; __bf16 e[8]; } vv;
      vv.q = *(const u4*)&qkv[base + 2048 + seg];
#pragma unroll
      for (int u = 0; u < 8; ++u) Vt[(seg + u) * LDV + row] = vv.e[u];
    }
    __syncthreads();

    // S = (Q K^T) * 0.125
    {
      v8f sacc[4][2] = {};
#pragma unroll
      for (int kk = 0; kk < 2; ++kk) {
        const int kh = kk * 32 + ((lane >> 4) << 3);
        v16bf qa[4], kv[2];
#pragma unroll
        for (int i = 0; i < 4; ++i)
          qa[i] = ldfrag(&Qs[(rS * 64 + i * 16 + (lane & 15)) * LDQ + kh]);
#pragma unroll
        for (int j = 0; j < 2; ++j)
          kv[j] = ldfrag(&Ks[(cS * 32 + j * 16 + (lane & 15)) * LDQ + kh]);
#pragma unroll
        for (int i = 0; i < 4; ++i)
#pragma unroll
          for (int j = 0; j < 2; ++j)
            sacc[i][j] = __builtin_amdgcn_wmma_f32_16x16x32_bf16(
                false, qa[i], false, kv[j], (short)0, sacc[i][j], false, false);
      }
#pragma unroll
      for (int i = 0; i < 4; ++i)
#pragma unroll
        for (int j = 0; j < 2; ++j) {
          const int col = cS * 32 + j * 16 + (lane & 15);
          const int rb = rS * 64 + i * 16 + ((lane >> 4) << 3);
#pragma unroll
          for (int rr = 0; rr < 8; ++rr)
            Sb[(rb + rr) * LDS_S + col] = sacc[i][j][rr] * 0.125f;
        }
    }
    __syncthreads();

    // Online softmax, one thread per row
    if (tid < 128) {
      float mx = -1e30f;
      for (int c = 0; c < 128; ++c) mx = fmaxf(mx, Sb[tid * LDS_S + c]);
      const float mold = mrow[tid];
      const float mnew = fmaxf(mold, mx);
      const float al = __expf(mold - mnew);
      float ssum = 0.f;
      for (int c = 0; c < 128; ++c) {
        const float p = __expf(Sb[tid * LDS_S + c] - mnew);
        Pb[tid * LDP + c] = (__bf16)p;
        ssum += p;
      }
      lrow[tid] = lrow[tid] * al + ssum;
      mrow[tid] = mnew;
      arow[tid] = al;
    }
    __syncthreads();

    // Rescale O, then O += P @ V^T
#pragma unroll
    for (int i = 0; i < 2; ++i)
#pragma unroll
      for (int j = 0; j < 2; ++j) {
        const int rb = rO * 32 + i * 16 + ((lane >> 4) << 3);
#pragma unroll
        for (int rr = 0; rr < 8; ++rr) oacc[i][j][rr] *= arow[rb + rr];
      }
#pragma unroll
    for (int kk = 0; kk < 4; ++kk) {
      const int kh = kk * 32 + ((lane >> 4) << 3);
      v16bf pa[2], vb[2];
#pragma unroll
      for (int i = 0; i < 2; ++i)
        pa[i] = ldfrag(&Pb[(rO * 32 + i * 16 + (lane & 15)) * LDP + kh]);
#pragma unroll
      for (int j = 0; j < 2; ++j)
        vb[j] = ldfrag(&Vt[(cO * 32 + j * 16 + (lane & 15)) * LDV + kh]);
#pragma unroll
      for (int i = 0; i < 2; ++i)
#pragma unroll
        for (int j = 0; j < 2; ++j)
          oacc[i][j] = __builtin_amdgcn_wmma_f32_16x16x32_bf16(
              false, pa[i], false, vb[j], (short)0, oacc[i][j], false, false);
    }
    __syncthreads();
  }

  // Normalize + write out (token-major, head block h*64)
#pragma unroll
  for (int i = 0; i < 2; ++i)
#pragma unroll
    for (int j = 0; j < 2; ++j) {
      const int d = cO * 32 + j * 16 + (lane & 15);
      const int rb = rO * 32 + i * 16 + ((lane >> 4) << 3);
#pragma unroll
      for (int rr = 0; rr < 8; ++rr) {
        const int r = rb + rr;
        const float v = oacc[i][j][rr] / lrow[r];
        out[(tokbase + qt * 128 + r) * 1024 + h * 64 + d] = (__bf16)v;
      }
    }
}

// ---------------------------------------------------------------------------
extern "C" void kernel_launch(void* const* d_in, const int* in_sizes, int n_in,
                              void* d_out, int out_size, void* d_ws,
                              size_t ws_size, hipStream_t stream) {
  (void)in_sizes; (void)n_in; (void)out_size; (void)ws_size;
  const float* x     = (const float*)d_in[0];
  const float* ln1g  = (const float*)d_in[1];
  const float* ln1b  = (const float*)d_in[2];
  const float* wqkv  = (const float*)d_in[3];
  const float* wproj = (const float*)d_in[4];
  const float* bproj = (const float*)d_in[5];
  const float* ln2g  = (const float*)d_in[6];
  const float* ln2b  = (const float*)d_in[7];
  const float* w1    = (const float*)d_in[8];
  const float* b1    = (const float*)d_in[9];
  const float* w2    = (const float*)d_in[10];
  const float* b2    = (const float*)d_in[11];
  float* out = (float*)d_out;

  char* ws = (char*)d_ws;
  const size_t MB = 1ull << 20;
  __bf16* h1     = (__bf16*)(ws + 0 * MB);    //  8 MB  LN1(x) bf16
  __bf16* qkv    = (__bf16*)(ws + 8 * MB);    // 24 MB  qkv bf16
  __bf16* aout   = (__bf16*)(ws + 32 * MB);   //  8 MB  attention out bf16
  float*  x1     = (float*)(ws + 40 * MB);    // 16 MB  x + proj f32
  __bf16* h2     = (__bf16*)(ws + 56 * MB);   //  8 MB  LN2(x1) bf16
  __bf16* h3     = (__bf16*)(ws + 64 * MB);   // 32 MB  GELU(mlp mid) bf16
  __bf16* wqkvT  = (__bf16*)(ws + 96 * MB);   //  6 MB
  __bf16* wprojT = (__bf16*)(ws + 102 * MB);  //  2 MB
  __bf16* w1T    = (__bf16*)(ws + 104 * MB);  //  8 MB
  __bf16* w2T    = (__bf16*)(ws + 112 * MB);  //  8 MB

  const dim3 tb(32, 8);
  trcvt_k<<<dim3(3072 / 32, 1024 / 32), tb, 0, stream>>>(wqkv, wqkvT, 1024, 3072);
  trcvt_k<<<dim3(1024 / 32, 1024 / 32), tb, 0, stream>>>(wproj, wprojT, 1024, 1024);
  trcvt_k<<<dim3(4096 / 32, 1024 / 32), tb, 0, stream>>>(w1, w1T, 1024, 4096);
  trcvt_k<<<dim3(1024 / 32, 4096 / 32), tb, 0, stream>>>(w2, w2T, 4096, 1024);

  ln_bf16_k<<<4096, 256, 0, stream>>>(x, ln1g, ln1b, h1);
  gemm_bf16_k<0><<<dim3(3072 / 128, 4096 / 128), 256, 0, stream>>>(
      h1, wqkvT, nullptr, nullptr, qkv, 4096, 3072, 1024);
  attn_k<<<dim3(16, 32), 256, 0, stream>>>(qkv, aout);
  gemm_bf16_k<1><<<dim3(1024 / 128, 4096 / 128), 256, 0, stream>>>(
      aout, wprojT, bproj, x, x1, 4096, 1024, 1024);
  ln_bf16_k<<<4096, 256, 0, stream>>>(x1, ln2g, ln2b, h2);
  gemm_bf16_k<2><<<dim3(4096 / 128, 4096 / 128), 256, 0, stream>>>(
      h2, w1T, b1, nullptr, h3, 4096, 4096, 1024);
  gemm_bf16_k<1><<<dim3(1024 / 128, 4096 / 128), 256, 0, stream>>>(
      h3, w2T, b2, x1, out, 4096, 1024, 4096);
}
